// QAttN_68015102099826
// MI455X (gfx1250) — compile-verified, matrix operand order checked by reading, not run on hardware
//
#include <hip/hip_runtime.h>
#include <math.h>
#include <stdint.h>

// ---------------- problem constants ----------------
#define Bc   32
#define Tc   50
#define Ec   100
#define G3   300          // 3*E
#define HIDc 64
#define NCc  7
#define BTc  (Bc * Tc)    // 1600
#define TP   64           // padded T (rows per (m,b) tile)
#define EP   128          // padded E (K dim for E-contractions)
#define MP   112          // padded measurement count (7*16)
#define SP   64           // padded s (K dim for att @ M)
#define KP0  320          // padded in-dims
#define KP1  96
#define KP2  64
#define G3P  304          // padded 3E rows (19*16)
#define FKP  320          // padded feat K (300 -> 320)

// ---------------- f32 workspace layout (float offsets) ----------------
static const size_t OFF_XI   = 0;                                    // 3*BT*G3
static const size_t OFF_REP  = OFF_XI  + (size_t)3 * BTc * G3;       // 3*BT*E
static const size_t OFF_W    = OFF_REP + (size_t)3 * BTc * Ec;       // 3*BT
static const size_t OFF_S    = OFF_W   + (size_t)3 * BTc;            // 6*32*64*64
static const size_t OFF_AR   = OFF_S   + (size_t)6 * Bc * TP * TP;   // 96*64*112 each
static const size_t OFF_BR   = OFF_AR  + (size_t)3 * Bc * TP * MP;
static const size_t OFF_AI   = OFF_BR  + (size_t)3 * Bc * TP * MP;
static const size_t OFF_BI   = OFF_AI  + (size_t)3 * Bc * TP * MP;
static const size_t OFF_LOG  = OFF_BI  + (size_t)3 * Bc * TP * MP;   // BT*NC
static const size_t F32_TOTAL = OFF_LOG + (size_t)BTc * NCc;         // multiple of 4

// ---------------- f16 arena layout (half offsets; all sizes multiple of 16) ----
static const size_t H_XH0  = 0;                                      // 1600*320
static const size_t H_XH1  = H_XH0  + (size_t)BTc * KP0;             // 1600*96
static const size_t H_XH2  = H_XH1  + (size_t)BTc * KP1;             // 1600*64
static const size_t H_WIH0 = H_XH2  + (size_t)BTc * KP2;             // 304*320
static const size_t H_WIH1 = H_WIH0 + (size_t)G3P * KP0;             // 304*96
static const size_t H_WIH2 = H_WIH1 + (size_t)G3P * KP1;             // 304*64
static const size_t H_WHH  = H_WIH2 + (size_t)G3P * KP2;             // 3*304*128
static const size_t H_PRH  = H_WHH  + (size_t)3 * G3P * EP;          // 96*64*128
static const size_t H_PIH  = H_PRH  + (size_t)3 * Bc * TP * EP;
static const size_t H_MRH  = H_PIH  + (size_t)3 * Bc * TP * EP;      // 112*128
static const size_t H_MIH  = H_MRH  + (size_t)MP * EP;
static const size_t H_ATT  = H_MIH  + (size_t)MP * EP;               // 96*64*64
static const size_t H_MT   = H_ATT  + (size_t)3 * Bc * TP * SP;      // 96*112*64
static const size_t H_FEAT = H_MT   + (size_t)3 * Bc * MP * SP;      // 1600*320
static const size_t H_W1   = H_FEAT + (size_t)BTc * FKP;             // 64*320
static const size_t H_W2   = H_W1   + (size_t)HIDc * FKP;            // 16*64
static const size_t H_H1   = H_W2   + (size_t)16 * 64;               // 1600*64
static const size_t H_TOTAL = H_H1  + (size_t)BTc * HIDc;            // even

// ---------------- WMMA helpers (gfx1250, wave32) ----------------
typedef __attribute__((ext_vector_type(16))) _Float16 v16h;
typedef __attribute__((ext_vector_type(8)))  _Float16 v8h;
typedef __attribute__((ext_vector_type(8)))  float    v8f;

#define DI __device__ __forceinline__

// Branchless fragment load from padded f16 row-major X[.][ldk]:
// lane<16 holds row r0+lane, K = k0+{0..7,16..23}; lane>=16: row r0+lane-16,
// K = k0+{8..15,24..31}. ldk multiple of 8 halves -> 16B-aligned b128 loads.
DI v16h load_frag16(const _Float16* X, int r0, int k0, int ldk, int lane) {
    const _Float16* p = X + (size_t)(r0 + (lane & 15)) * ldk + k0 + ((lane & 16) ? 8 : 0);
    v8h lo = *(const v8h*)(p);
    v8h hi = *(const v8h*)(p + 16);
    v16h f;
#pragma unroll
    for (int e = 0; e < 8; ++e) { f[e] = lo[e]; f[e + 8] = hi[e]; }
    return f;
}

DI v8f wmma16(v16h a, v16h b, v8f c) {
    return __builtin_amdgcn_wmma_f32_16x16x32_f16(
        false, a, false, b, (short)0, c, false, false);
}

DI float sigmoidf_(float x) { return 1.0f / (1.0f + expf(-x)); }

// gfx1250 async global->LDS copy (16B per lane), ASYNCcnt-tracked (ISA §10/§15.18)
DI void async_g2l_b128(uint32_t lds_addr, const _Float16* gaddr) {
    asm volatile("global_load_async_to_lds_b128 %0, %1, off"
                 :: "v"(lds_addr), "v"(gaddr) : "memory");
}
DI void wait_asynccnt0() {
    asm volatile("s_wait_asynccnt 0x0" ::: "memory");
}

// ---------------- arena zero-fill ----------------
__global__ void zero_u32_kernel(unsigned int* __restrict__ p, long long n) {
    long long i = (long long)blockIdx.x * blockDim.x + threadIdx.x;
    if (i < n) p[i] = 0u;
}

// ---------------- fused f32 -> padded f16 conversions (one launch) ----------------
#define NJOBS 13
struct CvtJobs {
    const float* src[NJOBS];
    _Float16*    dst[NJOBS];
    int cols[NJOBS];
    int ldd[NJOBS];
    int start[NJOBS + 1];
};
__global__ void cvt_pad_multi_kernel(CvtJobs J) {
    int i = blockIdx.x * blockDim.x + threadIdx.x;
    if (i >= J.start[NJOBS]) return;
    int j = 0;
    while (i >= J.start[j + 1]) ++j;
    int local = i - J.start[j];
    int r = local / J.cols[j];
    int c = local - r * J.cols[j];
    J.dst[j][(size_t)r * J.ldd[j] + c] = (_Float16)J.src[j][local];
}

// ---------------- generic GEMM: C = act(A[MxK,f16] * B[NxK,f16]^T + bias) -------
// B panel staged to LDS via async DMA, shared by 4 waves; A/B fragment loads are
// software-pipelined one k-step ahead so the wait before each WMMA is partial.
__global__ __launch_bounds__(128) void gemm_f16_kernel(
        float* __restrict__ C, _Float16* __restrict__ C16,
        const _Float16* __restrict__ A, const _Float16* __restrict__ Bh,
        const float* __restrict__ bias,
        int M, int N, int K, int ldc16, int act) {
    __shared__ __align__(16) _Float16 Btile[16 * FKP];   // K <= 320 -> 10 KB

    int tid = threadIdx.x, lane = tid & 31, wave = tid >> 5;
    int n0 = blockIdx.y * 16;

    // stage B panel (16 rows x K halves) into LDS with async copies
    int kg = K >> 3;                 // 16B chunks per row
    int chunks = 16 * kg;
    uint32_t lbase = (uint32_t)(uintptr_t)(&Btile[0]);
    for (int c = tid; c < chunks; c += 128) {
        int r = c / kg, cg = c - r * kg;
        async_g2l_b128(lbase + (uint32_t)((r * K + cg * 8) * 2),
                       Bh + (size_t)(n0 + r) * K + cg * 8);
    }
    wait_asynccnt0();
    __syncthreads();

    int m0 = (blockIdx.x * 4 + wave) * 16;
    if (m0 >= M) return;  // wave-uniform, after barrier

    const _Float16* Arow = A + (size_t)(m0 + (lane & 15)) * K;

    v8f acc = {};
    v16h fa = load_frag16(A, m0, 0, K, lane);
    v16h fb = load_frag16(Btile, 0, 0, K, lane);
    for (int k0 = 0; k0 < K; k0 += 32) {
        int kn  = k0 + 32;
        int kld = (kn < K) ? kn : k0;          // branchless clamp for the tail
        int kpf = (k0 + 64 < K) ? k0 + 64 : 0; // branchless prefetch index
        __builtin_prefetch(Arow + kpf, 0, 3);
        v16h fa2 = load_frag16(A, m0, kld, K, lane);       // next k-step (in flight
        v16h fb2 = load_frag16(Btile, 0, kld, K, lane);    //  during this WMMA)
        acc = wmma16(fa, fb, acc);
        fa = fa2; fb = fb2;
    }

    int n = n0 + (lane & 15);
    float bv = (bias && n < N) ? bias[n] : 0.0f;
    int mb = m0 + ((lane & 16) ? 8 : 0);
#pragma unroll
    for (int v = 0; v < 8; ++v) {
        int m = mb + v;
        float x = acc[v] + bv;
        if (act == 1)      x = fmaxf(x, 0.0f);
        else if (act == 2) x = tanhf(x);
        if (C && n < N)  C[(size_t)m * N + n] = x;
        if (C16)         C16[(size_t)m * ldc16 + n] = (_Float16)x;
    }
}

// ---------------- GRU scan: one workgroup per modality ----------------
__global__ __launch_bounds__(512) void gru_scan_kernel(
    float* __restrict__ rep_all, const float* __restrict__ xi_all,
    const _Float16* __restrict__ Whh_all,
    const float* __restrict__ bh0, const float* __restrict__ bh1, const float* __restrict__ bh2) {
    __shared__ float h_lds[Bc * Ec];                    // 12.8 KB
    __shared__ __align__(16) _Float16 h16[Bc * EP];     // 8 KB, zero-padded K
    __shared__ float gh_lds[Bc * G3];                   // 38.4 KB

    int m = blockIdx.x;
    const _Float16* Whh = Whh_all + (size_t)m * G3P * EP;
    const float* bh = (m == 0) ? bh0 : (m == 1) ? bh1 : bh2;
    const float* xi = xi_all + (size_t)m * BTc * G3;
    float* rep = rep_all + (size_t)m * BTc * Ec;

    int tid = threadIdx.x, lane = tid & 31, wave = tid >> 5;  // 16 waves

    for (int i = tid; i < Bc * Ec; i += 512) h_lds[i] = 0.0f;
    for (int i = tid; i < Bc * EP; i += 512) h16[i] = (_Float16)0.0f;
    __syncthreads();

    const int NT = 19;  // ceil(300/16); M tiles = 2
    for (int t = 0; t < Tc; ++t) {
        for (int tile = wave; tile < 2 * NT; tile += 16) {
            int m0 = (tile / NT) * 16;
            int n0 = (tile % NT) * 16;
            v8f acc = {};
#pragma unroll
            for (int k0 = 0; k0 < EP; k0 += 32) {
                v16h fa = load_frag16(h16, m0, k0, EP, lane);
                v16h fb = load_frag16(Whh, n0, k0, EP, lane);
                acc = wmma16(fa, fb, acc);
            }
            int n = n0 + (lane & 15);
            if (n < G3) {
                int mb = m0 + ((lane & 16) ? 8 : 0);
#pragma unroll
                for (int v = 0; v < 8; ++v)
                    gh_lds[(mb + v) * G3 + n] = acc[v] + bh[n];
            }
        }
        __syncthreads();

        for (int i = tid; i < Bc * Ec; i += 512) {
            int b = i / Ec, e = i % Ec;
            const float* xbt = xi + ((size_t)b * Tc + t) * G3;
            float r  = sigmoidf_(xbt[e]          + gh_lds[b * G3 + e]);
            float z  = sigmoidf_(xbt[Ec + e]     + gh_lds[b * G3 + Ec + e]);
            float nn = tanhf(xbt[2 * Ec + e] + r * gh_lds[b * G3 + 2 * Ec + e]);
            float h2 = (1.0f - z) * nn + z * h_lds[i];
            h_lds[i] = h2;
            h16[b * EP + e] = (_Float16)h2;
            rep[((size_t)b * Tc + t) * Ec + e] = tanhf(h2);
        }
        __syncthreads();
    }
}

// ---------------- norm / normalize / position phase -> w, prh, pih ----------------
__global__ void phase_kernel(_Float16* __restrict__ prh, _Float16* __restrict__ pih,
                             float* __restrict__ w_all, const float* __restrict__ rep_all,
                             const float* f0, const float* f1, const float* f2,
                             const float* c0, const float* c1, const float* c2) {
    int idx = blockIdx.x * blockDim.x + threadIdx.x;
    if (idx >= 3 * BTc) return;
    int m = idx / BTc, bt = idx % BTc;
    int b = bt / Tc, t = bt % Tc;
    const float* fr = (m == 0) ? f0 : (m == 1) ? f1 : f2;
    const float* bs = (m == 0) ? c0 : (m == 1) ? c1 : c2;
    const float* rep = rep_all + (size_t)m * BTc * Ec + (size_t)bt * Ec;

    float ss = 0.0f;
    for (int e = 0; e < Ec; ++e) { float v = rep[e]; ss += v * v; }
    float wn = sqrtf(ss);
    w_all[m * BTc + bt] = wn;
    float inv = 1.0f / fmaxf(wn, 1e-12f);

    size_t row = ((size_t)(m * Bc + b) * TP + t) * EP;
    float tf = (float)t;
    for (int e = 0; e < Ec; ++e) {
        float amp = rep[e] * inv;
        float ph = tf * fr[e] + bs[e];
        prh[row + e] = (_Float16)(cosf(ph) * amp);
        pih[row + e] = (_Float16)(sinf(ph) * amp);
    }
}

DI int other_of(int ind, int j) {
    if (ind == 0) return j == 0 ? 1 : 2;
    if (ind == 1) return j == 0 ? 0 : 2;
    return j == 0 ? 0 : 1;
}

// ---------------- pair scores: S_p(b,t,s); 4 waves = 4 s-tiles --------------------
// grid: (4 t-tiles, 6*B), block = 128
__global__ __launch_bounds__(128) void pair_score_kernel(
        float* __restrict__ S_all,
        const _Float16* __restrict__ prh, const _Float16* __restrict__ pih) {
    int lane = threadIdx.x & 31, wave = threadIdx.x >> 5;
    int t0 = blockIdx.x * 16, s0 = wave * 16;
    int z = blockIdx.y;
    int p = z / Bc, b = z % Bc;
    int ind = p >> 1;
    int o = other_of(ind, p & 1);

    const _Float16* pro = prh + (size_t)(o * Bc + b) * TP * EP;
    const _Float16* pio = pih + (size_t)(o * Bc + b) * TP * EP;
    const _Float16* av  = prh + (size_t)(ind * Bc + b) * TP * EP;
    const _Float16* bv  = pih + (size_t)(ind * Bc + b) * TP * EP;

    v8f ra = {}, ia = {}, rb = {}, ib = {};
#pragma unroll
    for (int k0 = 0; k0 < EP; k0 += 32) {
        v16h fra = load_frag16(pro, t0, k0, EP, lane);
        v16h fia = load_frag16(pio, t0, k0, EP, lane);
        v16h fav = load_frag16(av,  s0, k0, EP, lane);
        v16h fbv = load_frag16(bv,  s0, k0, EP, lane);
        ra = wmma16(fra, fav, ra);
        ia = wmma16(fia, fav, ia);
        rb = wmma16(fra, fbv, rb);
        ib = wmma16(fia, fbv, ib);
    }

    float* S = S_all + (size_t)z * TP * TP;
    int s = s0 + (lane & 15);
    int tb = t0 + ((lane & 16) ? 8 : 0);
#pragma unroll
    for (int v = 0; v < 8; ++v) {
        float gra = ra[v], gia = ia[v], grb = rb[v], gib = ib[v];
        S[(tb + v) * TP + s] = gra * gra + gia * gia + grb * grb + gib * gib
                             + 2.0f * (gib * gra - grb * gia);
    }
}

// ---------------- batched projections (all 4 combos in one launch) ----------------
// grid: (7 n-tiles, 96 mb, 4 combos), block = 128 (wave = m-tile)
__global__ __launch_bounds__(128) void bproj_kernel(
        float* __restrict__ Ar, float* __restrict__ Br,
        float* __restrict__ Ai, float* __restrict__ Bi,
        const _Float16* __restrict__ prh, const _Float16* __restrict__ pih,
        const _Float16* __restrict__ mrh, const _Float16* __restrict__ mih) {
    int lane = threadIdx.x & 31, wave = threadIdx.x >> 5;
    int n0 = blockIdx.x * 16;
    int mb = blockIdx.y;
    int q  = blockIdx.z;
    const _Float16* P  = (q & 1)  ? pih : prh;
    const _Float16* Mh = (q >> 1) ? mih : mrh;
    float* Cq = (q == 0) ? Ar : (q == 1) ? Br : (q == 2) ? Ai : Bi;

    const _Float16* A = P + (size_t)mb * TP * EP;
    int m0 = wave * 16;
    v8f acc = {};
#pragma unroll
    for (int k0 = 0; k0 < EP; k0 += 32) {
        v16h fa = load_frag16(A,  m0, k0, EP, lane);
        v16h fb = load_frag16(Mh, n0, k0, EP, lane);
        acc = wmma16(fa, fb, acc);
    }
    float* Cp = Cq + (size_t)mb * TP * MP;
    int n = n0 + (lane & 15);
    int tb = m0 + ((lane & 16) ? 8 : 0);
#pragma unroll
    for (int v = 0; v < 8; ++v) Cp[(tb + v) * MP + n] = acc[v];
}

// ---------------- M^T(b,k,s) from the 4 projections ----------------
__global__ void mvec_kernel(_Float16* __restrict__ MT, const float* __restrict__ Ar,
                            const float* __restrict__ Br, const float* __restrict__ Ai,
                            const float* __restrict__ Bi) {
    int i = blockIdx.x * blockDim.x + threadIdx.x;
    if (i >= 3 * Bc * Ec * Tc) return;
    int t = i % Tc;
    int k = (i / Tc) % Ec;
    int mb = i / (Tc * Ec);
    size_t src = ((size_t)mb * TP + t) * MP + k;
    float ar = Ar[src], br = Br[src], ai = Ai[src], bi = Bi[src];
    float val = ar * ar + br * br + ai * ai + bi * bi + 2.0f * (bi * ar - ai * br);
    MT[((size_t)mb * MP + k) * SP + t] = (_Float16)val;
}

// ---------------- mixture weights + score softmax -> att ----------------
__global__ void attn_kernel(_Float16* __restrict__ atth, const float* __restrict__ S_all,
                            const float* __restrict__ w_all) {
    int idx = blockIdx.x * blockDim.x + threadIdx.x;
    if (idx >= 3 * BTc) return;
    int ind = idx / BTc, bt = idx % BTc;
    int b = bt / Tc, t = bt % Tc;
    int o1 = other_of(ind, 0), o2 = other_of(ind, 1);
    float w1 = w_all[o1 * BTc + bt], w2 = w_all[o2 * BTc + bt];
    float mx = fmaxf(w1, w2);
    float e1 = expf(w1 - mx), e2 = expf(w2 - mx);
    float inv12 = 1.0f / (e1 + e2);
    float mw1 = e1 * inv12, mw2 = e2 * inv12;

    const float* S1 = S_all + ((size_t)((2 * ind) * Bc + b) * TP + t) * TP;
    const float* S2 = S_all + ((size_t)((2 * ind + 1) * Bc + b) * TP + t) * TP;
    const float* kw = w_all + (size_t)ind * BTc + (size_t)b * Tc;

    float sc[Tc];
    float smax = -3.4e38f;
    for (int s = 0; s < Tc; ++s) {
        float v = (mw1 * S1[s] + mw2 * S2[s]) * kw[s];
        sc[s] = v;
        smax = fmaxf(smax, v);
    }
    float den = 0.0f;
    for (int s = 0; s < Tc; ++s) { float e = expf(sc[s] - smax); sc[s] = e; den += e; }
    float inv = 1.0f / den;
    _Float16* att = atth + ((size_t)(ind * Bc + b) * TP + t) * SP;
    for (int s = 0; s < Tc; ++s) att[s] = (_Float16)(sc[s] * inv);
}

// ---------------- feat = att @ M : NT WMMA against transposed M ----------------
// grid: (7 k-tiles, 3*B), block = 128 (wave = t-tile)
__global__ __launch_bounds__(128) void att_apply_kernel(
        _Float16* __restrict__ feath,
        const _Float16* __restrict__ atth, const _Float16* __restrict__ MT) {
    int lane = threadIdx.x & 31, wave = threadIdx.x >> 5;
    int n0 = blockIdx.x * 16;
    int mb = blockIdx.y;
    int ind = mb / Bc, b = mb % Bc;
    int t0 = wave * 16;

    const _Float16* A  = atth + (size_t)mb * TP * SP;  // rows t, K = s
    const _Float16* Bp = MT   + (size_t)mb * MP * SP;  // rows k, K = s

    v8f acc = {};
#pragma unroll
    for (int k0 = 0; k0 < SP; k0 += 32) {
        v16h fa = load_frag16(A,  t0, k0, SP, lane);
        v16h fb = load_frag16(Bp, n0, k0, SP, lane);
        acc = wmma16(fa, fb, acc);
    }

    int n = n0 + (lane & 15);
    if (n < Ec) {
        int tb = t0 + ((lane & 16) ? 8 : 0);
#pragma unroll
        for (int v = 0; v < 8; ++v) {
            int t = tb + v;
            if (t < Tc)
                feath[((size_t)b * Tc + t) * FKP + ind * Ec + n] = (_Float16)acc[v];
        }
    }
}

// ---------------- log-softmax over NC=7 ----------------
__global__ void logsoftmax_kernel(float* __restrict__ out, const float* __restrict__ logits) {
    int i = blockIdx.x * blockDim.x + threadIdx.x;
    if (i >= BTc) return;
    const float* l = logits + (size_t)i * NCc;
    float mx = -3.4e38f;
    for (int k = 0; k < NCc; ++k) mx = fmaxf(mx, l[k]);
    float den = 0.0f;
    for (int k = 0; k < NCc; ++k) den += expf(l[k] - mx);
    float lden = logf(den);
    for (int k = 0; k < NCc; ++k) out[(size_t)i * NCc + k] = l[k] - mx - lden;
}

// ---------------- host-side orchestration ----------------
extern "C" void kernel_launch(void* const* d_in, const int* in_sizes, int n_in,
                              void* d_out, int out_size, void* d_ws, size_t ws_size,
                              hipStream_t stream) {
    (void)in_sizes; (void)n_in; (void)out_size; (void)ws_size;
    const int DIMS[3] = {300, 74, 35};
    const int KPAD[3] = {KP0, KP1, KP2};

    const float* x[3];  const float* Wi[3]; const float* Wh[3];
    const float* bi[3]; const float* bh[3]; const float* fr[3]; const float* bs[3];
    for (int m = 0; m < 3; ++m) {
        int base = 7 * m;
        x[m]  = (const float*)d_in[base + 0];
        Wi[m] = (const float*)d_in[base + 1];
        Wh[m] = (const float*)d_in[base + 2];
        bi[m] = (const float*)d_in[base + 3];
        bh[m] = (const float*)d_in[base + 4];
        fr[m] = (const float*)d_in[base + 5];
        bs[m] = (const float*)d_in[base + 6];
    }
    const float* meas_r = (const float*)d_in[21];
    const float* meas_i = (const float*)d_in[22];
    const float* W1 = (const float*)d_in[23];
    const float* b1 = (const float*)d_in[24];
    const float* W2 = (const float*)d_in[25];
    const float* b2 = (const float*)d_in[26];

    float* ws = (float*)d_ws;
    float* xi     = ws + OFF_XI;
    float* rep    = ws + OFF_REP;
    float* w      = ws + OFF_W;
    float* S      = ws + OFF_S;
    float* Ar     = ws + OFF_AR;
    float* Br     = ws + OFF_BR;
    float* Ai     = ws + OFF_AI;
    float* Bi     = ws + OFF_BI;
    float* logits = ws + OFF_LOG;

    _Float16* hb   = (_Float16*)(ws + F32_TOTAL);
    _Float16* xh[3]  = { hb + H_XH0,  hb + H_XH1,  hb + H_XH2  };
    _Float16* Wih[3] = { hb + H_WIH0, hb + H_WIH1, hb + H_WIH2 };
    _Float16* Whh  = hb + H_WHH;
    _Float16* prh  = hb + H_PRH;
    _Float16* pih  = hb + H_PIH;
    _Float16* mrh  = hb + H_MRH;
    _Float16* mih  = hb + H_MIH;
    _Float16* atth = hb + H_ATT;
    _Float16* MT   = hb + H_MT;
    _Float16* feath= hb + H_FEAT;
    _Float16* W1h  = hb + H_W1;
    _Float16* W2h  = hb + H_W2;
    _Float16* h1h  = hb + H_H1;

    // 0) zero f16 arena (padding must read as 0 on every replay)
    {
        long long n32 = (long long)(H_TOTAL / 2);
        zero_u32_kernel<<<(unsigned)((n32 + 255) / 256), 256, 0, stream>>>((unsigned int*)hb, n32);
    }

    // 1) all f32->f16 padded conversions in a single launch
    {
        CvtJobs J;
        int j = 0, acc = 0;
        auto add = [&](const float* s, _Float16* d, int rows, int cols, int ldd) {
            J.src[j] = s; J.dst[j] = d; J.cols[j] = cols; J.ldd[j] = ldd;
            J.start[j] = acc; acc += rows * cols; ++j;
        };
        for (int m = 0; m < 3; ++m) add(x[m],  xh[m],  BTc, DIMS[m], KPAD[m]);
        for (int m = 0; m < 3; ++m) add(Wi[m], Wih[m], G3,  DIMS[m], KPAD[m]);
        for (int m = 0; m < 3; ++m) add(Wh[m], Whh + (size_t)m * G3P * EP, G3, Ec, EP);
        add(meas_r, mrh, Ec, Ec, EP);
        add(meas_i, mih, Ec, Ec, EP);
        add(W1, W1h, HIDc, G3, FKP);
        add(W2, W2h, NCc, HIDc, HIDc);
        J.start[NJOBS] = acc;
        cvt_pad_multi_kernel<<<(acc + 255) / 256, 256, 0, stream>>>(J);
    }

    // 2) xi = x @ Wi^T + bi   (M=1600, N=300, K=KPAD[m])
    for (int m = 0; m < 3; ++m) {
        dim3 g(BTc / 64, G3P / 16);
        gemm_f16_kernel<<<g, 128, 0, stream>>>(xi + (size_t)m * BTc * G3, nullptr,
                                               xh[m], Wih[m], bi[m],
                                               BTc, G3, KPAD[m], 0, 0);
    }

    // 3) GRU scan (rep = tanh(h_t))
    gru_scan_kernel<<<3, 512, 0, stream>>>(rep, xi, Whh, bh[0], bh[1], bh[2]);

    // 4) norms, amps, phases -> w, prh, pih
    phase_kernel<<<(3 * BTc + 255) / 256, 256, 0, stream>>>(prh, pih, w, rep,
                                                            fr[0], fr[1], fr[2],
                                                            bs[0], bs[1], bs[2]);

    // 5) per-pair attention scores
    pair_score_kernel<<<dim3(TP / 16, Bc * 6), 128, 0, stream>>>(S, prh, pih);

    // 6) measurement projections (4 combos, one launch) + M^T
    bproj_kernel<<<dim3(MP / 16, 3 * Bc, 4), 128, 0, stream>>>(Ar, Br, Ai, Bi,
                                                               prh, pih, mrh, mih);
    mvec_kernel<<<(3 * Bc * Ec * Tc + 255) / 256, 256, 0, stream>>>(MT, Ar, Br, Ai, Bi);

    // 7) mixture + softmax -> att ; feat = att @ M
    attn_kernel<<<(3 * BTc + 255) / 256, 256, 0, stream>>>(atth, S, w);
    att_apply_kernel<<<dim3(MP / 16, 3 * Bc), 128, 0, stream>>>(feath, atth, MT);

    // 8) MLP + log_softmax
    {
        dim3 g1(BTc / 64, HIDc / 16);
        gemm_f16_kernel<<<g1, 128, 0, stream>>>(nullptr, h1h, feath, W1h, b1,
                                                BTc, HIDc, FKP, HIDc, 1);    // ReLU, f16 out
        dim3 g2(BTc / 64, 1);
        gemm_f16_kernel<<<g2, 128, 0, stream>>>(logits, nullptr, h1h, W2h, b2,
                                                BTc, NCc, HIDc, 0, 2);       // tanh
    }
    logsoftmax_kernel<<<(BTc + 255) / 256, 256, 0, stream>>>((float*)d_out, logits);
}
